// GraphAttention_34462817583586
// MI455X (gfx1250) — matrix operand order repaired and checked by reading, not compiled, not run on hardware
//
#include <hip/hip_runtime.h>
#include <hip/hip_bf16.h>
#include <stdint.h>

// ---------------------------------------------------------------------------
// GraphAttention fused kernel for gfx1250 (MI455X).
//   N=64, T=512, VN=10, C=128, HEADS=8, DH=16, INNER=128
//   b = N*T = 32768 batch items, ROWS = b*VN = 327680 rows of 128 channels.
// ---------------------------------------------------------------------------

#define C_DIM     128
#define VN_DIM    10
#define NROWS     327680          // 32768 * 10
#define QKV_COLS  384
#define BI_PER_BLK 8              // batch items per workgroup
#define ROWS_BLK  (BI_PER_BLK * VN_DIM)   // 80 rows per workgroup
#define YS        136             // f16 LDS row stride for y / attn-out (128 + 8 pad)
#define QS        392             // f32 LDS row stride for qkv (384 + 8 pad)
#define BN_EPS    1e-5f

typedef _Float16 v16h __attribute__((ext_vector_type(16)));
typedef _Float16 h4   __attribute__((ext_vector_type(4)));
typedef float    v8f  __attribute__((ext_vector_type(8)));
typedef float    f32x4 __attribute__((ext_vector_type(4)));
typedef uint32_t u32x4 __attribute__((ext_vector_type(4)));

union FragA { v16h v; u32x4 q[2]; };

// Dynamic LDS layout (bytes):
//   [0)      float sscale[128]
//   [512)    float sshift[128]
//   [1024)   float sbout [128]
//   [1536)   float smask [104]   (100 used)
//   [1952)   _Float16 y_s [80*136]   = 21760 B   (reused for attn output, f16)
//   [23712)  float qkv_s  [80*392]   = 125440 B  (phase A: aliased as x staging)
#define Y_OFF      1952
#define QKV_OFF    (Y_OFF + ROWS_BLK * YS * 2)            // 23712
#define SMEM_BYTES (QKV_OFF + ROWS_BLK * QS * 4)          // 149152

// ---------------------------------------------------------------------------
// Prep kernels
// ---------------------------------------------------------------------------
__global__ void ga_zero_stats(float* stats) {
    stats[threadIdx.x] = 0.0f;   // 256 threads: sum[128] + sumsq[128]
}

__global__ __launch_bounds__(256)
void ga_bn_stats(const float* __restrict__ x, float* __restrict__ stats) {
    __shared__ float sh[256];
    __shared__ float sh2[256];
    const int tid  = threadIdx.x;
    const int c    = tid & 127;
    const int half = tid >> 7;
    float s = 0.0f, ss = 0.0f;
    for (int r = blockIdx.x * 2 + half; r < NROWS; r += gridDim.x * 2) {
        float v = x[(size_t)r * C_DIM + c];
        s += v;
        ss += v * v;
    }
    sh[tid] = s;
    sh2[tid] = ss;
    __syncthreads();
    if (half == 0) {
        s  += sh[tid + 128];
        ss += sh2[tid + 128];
        atomicAdd(&stats[c], s);
        atomicAdd(&stats[128 + c], ss);
    }
}

__global__ void ga_bn_finalize(const float* __restrict__ stats,
                               const float* __restrict__ gamma,
                               const float* __restrict__ beta,
                               float* __restrict__ scaleshift) {
    const int c = threadIdx.x;     // 128 threads
    const float inv_n = 1.0f / (float)NROWS;
    float mean = stats[c] * inv_n;
    float var  = stats[128 + c] * inv_n - mean * mean;   // biased (matches BN)
    float sc   = gamma[c] * rsqrtf(var + BN_EPS);
    scaleshift[c]       = sc;
    scaleshift[128 + c] = beta[c] - mean * sc;
}

// Convert fp32 weights to f16, transposed to [n][k] so WMMA B fragments are
// contiguous 16-byte loads (K runs along memory).
__global__ __launch_bounds__(256)
void ga_conv_weights(const float* __restrict__ wqkv, const float* __restrict__ wout,
                     _Float16* __restrict__ wqkv_t, _Float16* __restrict__ wout_t) {
    const int idx = blockIdx.x * 256 + threadIdx.x;   // 65536 total
    if (idx < C_DIM * QKV_COLS) {                     // 49152
        int k = idx / QKV_COLS;
        int n = idx % QKV_COLS;
        wqkv_t[n * C_DIM + k] = (_Float16)wqkv[idx];
    } else {
        int i2 = idx - C_DIM * QKV_COLS;              // < 16384
        int k = i2 >> 7;
        int n = i2 & 127;
        wout_t[n * C_DIM + k] = (_Float16)wout[i2];
    }
}

// ---------------------------------------------------------------------------
// Fused main kernel: async x->LDS, BN affine -> QKV WMMA -> masked attention
// -> out-proj WMMA
// ---------------------------------------------------------------------------
__global__ __launch_bounds__(256)
void ga_fused(const float* __restrict__ x,
              const float* __restrict__ mask,
              const float* __restrict__ bout,
              const float* __restrict__ scale,
              const float* __restrict__ shift,
              const _Float16* __restrict__ wqkv_t,
              const _Float16* __restrict__ wout_t,
              float* __restrict__ out) {
    extern __shared__ char smem_raw[];
    float*    fp     = (float*)smem_raw;
    float*    sscale = fp;             // 128
    float*    sshift = fp + 128;       // 128
    float*    sbout  = fp + 256;       // 128
    float*    smask  = fp + 384;       // 100 (+4 pad)
    _Float16* y_s    = (_Float16*)(smem_raw + Y_OFF);    // 80*YS f16
    float*    qkv_s  = (float*)(smem_raw + QKV_OFF);     // 80*QS f32 (stage alias)

    const int tid     = threadIdx.x;
    const int lane    = tid & 31;
    const int wave    = tid >> 5;
    const int m16     = lane & 15;     // A-row / B-col / D-col within tile
    const int kh      = lane >> 4;     // lane-half selector
    const int rowbase = blockIdx.x * ROWS_BLK;

    // LDS byte address of the dynamic shared base (for async-to-LDS vdst).
    const uint32_t lds_base =
        (uint32_t)(uintptr_t)(__attribute__((address_space(3))) char*)smem_raw;

    // ---- Phase A0: async-copy raw f32 x tile into staging LDS (qkv_s area) --
    {
        const uint64_t gbase = (uint64_t)(uintptr_t)(x + (size_t)rowbase * C_DIM);
        for (int i = tid; i < ROWS_BLK * C_DIM / 4; i += 256) {   // 2560 B128 ops
            uint32_t voff  = (uint32_t)i * 16u;                   // global byte off
            uint32_t laddr = lds_base + (uint32_t)QKV_OFF + (uint32_t)i * 16u;
            asm volatile("global_load_async_to_lds_b128 %0, %1, %2"
                         :: "v"(laddr), "v"(voff), "s"(gbase)
                         : "memory");
        }
    }

    // ---- load per-channel constants + mask (overlaps async copies) ---------
    if (tid < 128) {
        sscale[tid] = scale[tid];
        sshift[tid] = shift[tid];
        sbout[tid]  = bout[tid];
    }
    if (tid < 100) smask[tid] = mask[tid];

    asm volatile("s_wait_asynccnt 0x0" ::: "memory");
    __syncthreads();

    // ---- Phase A1: staging f32 -> BN affine -> f16 y_s ---------------------
    {
        const float* stage = qkv_s;                       // aliased staging
        for (int i = tid; i < ROWS_BLK * C_DIM / 4; i += 256) {
            int r = i >> 5;                               // 32 float4 per row
            int c = (i & 31) * 4;
            f32x4 v = *(const f32x4*)(stage + i * 4);
            h4 hv;
            hv.x = (_Float16)(v.x * sscale[c + 0] + sshift[c + 0]);
            hv.y = (_Float16)(v.y * sscale[c + 1] + sshift[c + 1]);
            hv.z = (_Float16)(v.z * sscale[c + 2] + sshift[c + 2]);
            hv.w = (_Float16)(v.w * sscale[c + 3] + sshift[c + 3]);
            *(h4*)(y_s + r * YS + c) = hv;
        }
    }
    __syncthreads();

    // ---- Phase B: qkv = y @ Wqkv  (80x384, f32 acc) ------------------------
    // 5 M-tiles x 24 N-tiles = 120 tiles; 8 waves -> 15 tiles each; K=128 -> 4 WMMAs.
    for (int t = wave; t < 120; t += 8) {
        const int mt = t / 24;
        const int nt = t % 24;
        v8f acc = {0.f, 0.f, 0.f, 0.f, 0.f, 0.f, 0.f, 0.f};
        const _Float16* yrow = y_s + (mt * 16 + m16) * YS;
        const _Float16* bcol = wqkv_t + (nt * 16 + m16) * C_DIM;
#pragma unroll
        for (int kb = 0; kb < 4; ++kb) {
            const int kbase = kb * 32;
            FragA a, b;
            a.q[0] = *(const u32x4*)(yrow + kbase + kh * 8);        // K: kh*8+0..7
            a.q[1] = *(const u32x4*)(yrow + kbase + 16 + kh * 8);   // K: 16+kh*8+0..7
            b.q[0] = *(const u32x4*)(bcol + kbase + kh * 8);        // K: kh*8+0..7
            b.q[1] = *(const u32x4*)(bcol + kbase + 16 + kh * 8);   // K: 16+kh*8+0..7
            acc = __builtin_amdgcn_wmma_f32_16x16x32_f16(
                false, a.v, false, b.v, (short)0, acc, false, false);
        }
#pragma unroll
        for (int v = 0; v < 8; ++v)
            qkv_s[(mt * 16 + v + kh * 8) * QS + nt * 16 + m16] = acc[v];
    }
    __syncthreads();

    // ---- Phase C: per-(batch,head,row) masked attention --------------------
    // 8 bi * 8 heads * 10 rows = 640 query rows.
    for (int p = tid; p < BI_PER_BLK * 8 * VN_DIM; p += 256) {
        const int bi  = p / 80;
        const int rem = p % 80;
        const int h   = rem / 10;
        const int i   = rem % 10;
        const int r   = bi * VN_DIM + i;
        const float* qrow = qkv_s + r * QS + h * 16;

        float dots[VN_DIM];
        float mx = -1e30f;
#pragma unroll
        for (int j = 0; j < VN_DIM; ++j) {
            const float* krow = qkv_s + (bi * VN_DIM + j) * QS + 128 + h * 16;
            float s = 0.0f;
#pragma unroll
            for (int d = 0; d < 16; ++d) s += qrow[d] * krow[d];
            s = s * 0.25f * smask[i * VN_DIM + j];   // scale * multiplicative mask
            dots[j] = s;
            mx = fmaxf(mx, s);
        }
        float den = 0.0f;
#pragma unroll
        for (int j = 0; j < VN_DIM; ++j) {
            dots[j] = __expf(dots[j] - mx);
            den += dots[j];
        }
        const float inv = 1.0f / den;
        float o[16];
#pragma unroll
        for (int d = 0; d < 16; ++d) o[d] = 0.0f;
#pragma unroll
        for (int j = 0; j < VN_DIM; ++j) {
            const float w = dots[j] * inv;
            const float* vrow = qkv_s + (bi * VN_DIM + j) * QS + 256 + h * 16;
#pragma unroll
            for (int d = 0; d < 16; ++d) o[d] += w * vrow[d];
        }
        // write attention output (f16) over the y buffer
#pragma unroll
        for (int d = 0; d < 16; ++d)
            y_s[r * YS + h * 16 + d] = (_Float16)o[d];
    }
    __syncthreads();

    // ---- Phase D: out = ao @ Wout + b_out  (80x128) ------------------------
    // 5 M-tiles x 8 N-tiles = 40 tiles; 8 waves -> 5 each; K=128 -> 4 WMMAs.
    for (int t = wave; t < 40; t += 8) {
        const int mt = t / 8;
        const int nt = t % 8;
        v8f acc = {0.f, 0.f, 0.f, 0.f, 0.f, 0.f, 0.f, 0.f};
        const _Float16* arow = y_s + (mt * 16 + m16) * YS;
        const _Float16* bcol = wout_t + (nt * 16 + m16) * C_DIM;
#pragma unroll
        for (int kb = 0; kb < 4; ++kb) {
            const int kbase = kb * 32;
            FragA a, b;
            a.q[0] = *(const u32x4*)(arow + kbase + kh * 8);
            a.q[1] = *(const u32x4*)(arow + kbase + 16 + kh * 8);
            b.q[0] = *(const u32x4*)(bcol + kbase + kh * 8);
            b.q[1] = *(const u32x4*)(bcol + kbase + 16 + kh * 8);
            acc = __builtin_amdgcn_wmma_f32_16x16x32_f16(
                false, a.v, false, b.v, (short)0, acc, false, false);
        }
        const int col = nt * 16 + m16;
        const float bo = sbout[col];
#pragma unroll
        for (int v = 0; v < 8; ++v) {
            int R = rowbase + mt * 16 + v + kh * 8;
            out[(size_t)R * C_DIM + col] = acc[v] + bo;
        }
    }
}

// ---------------------------------------------------------------------------
// Host-side launcher
// ---------------------------------------------------------------------------
extern "C" void kernel_launch(void* const* d_in, const int* in_sizes, int n_in,
                              void* d_out, int out_size, void* d_ws, size_t ws_size,
                              hipStream_t stream) {
    const float* x     = (const float*)d_in[0];
    const float* gamma = (const float*)d_in[1];
    const float* beta  = (const float*)d_in[2];
    const float* wqkv  = (const float*)d_in[3];
    const float* wout  = (const float*)d_in[4];
    const float* bout  = (const float*)d_in[5];
    const float* mask  = (const float*)d_in[6];
    float* out = (float*)d_out;

    char* ws = (char*)d_ws;
    float*    stats      = (float*)ws;                 // 256 f32
    float*    scaleshift = (float*)(ws + 1024);        // 256 f32
    _Float16* wqkv_t     = (_Float16*)(ws + 2048);     // 384x128 f16
    _Float16* wout_t     = (_Float16*)(ws + 2048 + 98304); // 128x128 f16

    ga_zero_stats<<<1, 256, 0, stream>>>(stats);
    ga_bn_stats<<<1024, 256, 0, stream>>>(x, stats);
    ga_bn_finalize<<<1, 128, 0, stream>>>(stats, gamma, beta, scaleshift);
    ga_conv_weights<<<256, 256, 0, stream>>>(wqkv, wout, wqkv_t, wout_t);

    const int nblocks = NROWS / ROWS_BLK;   // 4096
    ga_fused<<<nblocks, 256, SMEM_BYTES, stream>>>(
        x, mask, bout, scaleshift, scaleshift + 128, wqkv_t, wout_t, out);
}
